// DDSP_7318624272846
// MI455X (gfx1250) — matrix operand order repaired and checked by reading, not compiled, not run on hardware
//
#include <hip/hip_runtime.h>
#include <hip/hip_bf16.h>

// ---- problem constants (from reference) ----
#define B_    32
#define CH_   64
#define BAND_ 16384
#define NC_   129      // NOISE_COEFFS
#define CHUNKS_ 128    // BAND / NOISE_STEP, also frame count
#define LDSTRIDE 132   // padded fp32 LDS row stride (row*132*4 is 16B aligned)
#define HSTRIDE  80    // padded f16 LDS row stride (row*80*2 = 160B, 16B aligned)

typedef __attribute__((ext_vector_type(16))) _Float16 v16h;
typedef __attribute__((ext_vector_type(8)))  float    v8f;

union FragA { v16h v; _Float16 h[16]; };
union FragD { v8f  v; float     f[8]; };

// =====================================================================
// Kernel 1: per (batch, 128-sample chunk):
//   - async-stage x[b, 0:64, t0:t0+128] (fp32) directly into LDS
//     via GLOBAL_LOAD_ASYNC_TO_LDS_B128 (ASYNCcnt)
//   - channel mean -> out[b, t]
//   - one cooperative f32->f16 transpose pass into a K-contiguous layout
//   - WMMA GEMM: clip(noise_w @ x + noise_b, 0, 1), mean over t -> spec
// grid = 32*128 blocks, 256 threads (8 waves; wave w owns M-tiles w, w+8)
// =====================================================================
__global__ __launch_bounds__(256) void ddsp_gemm_mean(
    const float* __restrict__ x,      // (32, 64, 16384)
    const float* __restrict__ nw,     // (129, 64) row-major
    const float* __restrict__ nb,     // (129,)
    float* __restrict__ out,          // (32, 16384)  <- channel mean written here
    float* __restrict__ spec)         // (32, 129, 128) workspace
{
    __shared__ __align__(16) float    xs[CH_ * LDSTRIDE];     // fp32 [k][t]
    __shared__ __align__(16) _Float16 xh[128 * HSTRIDE];      // f16  [t][k]

    const int blk  = blockIdx.x;
    const int b    = blk >> 7;     // / 128
    const int j    = blk & 127;
    const int t0   = j * 128;
    const int tid  = threadIdx.x;
    const int lane = tid & 31;
    const int wave = tid >> 5;

    // ---- async stage: each wave moves one 512B row (32 lanes x b128) per iter ----
    {
        const unsigned long long xbase = (unsigned long long)(uintptr_t)x;
        const unsigned base_off =
            (unsigned)((((size_t)b * CH_) * BAND_ + t0) * sizeof(float));
#pragma unroll
        for (int it = 0; it < 8; ++it) {
            const int row = wave + it * 8;
            const unsigned goff = base_off + (unsigned)row * (BAND_ * 4u) + lane * 16u;
            const unsigned loff =
                (unsigned)(uintptr_t)(&xs[row * LDSTRIDE + lane * 4]);
            asm volatile("global_load_async_to_lds_b128 %0, %1, %2 offset:0"
                         :: "v"(loff), "v"(goff), "s"(xbase) : "memory");
        }
        asm volatile("s_wait_asynccnt 0x0" ::: "memory");
    }
    __syncthreads();

    // ---- channel mean (fp32, exact path of the output) ----
    if (tid < 128) {
        float s = 0.f;
#pragma unroll 8
        for (int c = 0; c < CH_; ++c) s += xs[c * LDSTRIDE + tid];
        out[(size_t)b * BAND_ + t0 + tid] = s * (1.f / 64.f);
    }

    // ---- one-shot f32->f16 transpose: xh[t][k] = (f16) xs[k][t] ----
    {
        const int t  = tid >> 1;
        const int k0 = (tid & 1) * 32;
        _Float16* dst = &xh[t * HSTRIDE + k0];
#pragma unroll 8
        for (int e = 0; e < 32; ++e)
            dst[e] = (_Float16)xs[(k0 + e) * LDSTRIDE + t];
    }
    __syncthreads();

    // ---- WMMA GEMM: M = 129 coeffs (9 tiles of 16), N = 128 t (8 tiles), K = 64 ----
    const int mloc = lane & 15;       // M within tile (A row; also B/D column N)
    const int half = lane >> 4;       // lane half selects K(A)/M-offset(D)

    for (int mt = wave; mt < 9; mt += 8) {
        // Build A fragments (16x32 f16) for K blocks [0,32) and [32,64).
        // Contiguous K order (same as B): lane l -> M = mt*16 + (l&15); element e: K = 16*(l>>4) + e
        FragA a0, a1;
        const int mrow = mt * 16 + mloc;
#pragma unroll
        for (int e = 0; e < 16; ++e) {
            const int v = e >> 1, pos = e & 1;
            const int k = 16 * half + e; (void)v; (void)pos;
            float w0 = 0.f, w1 = 0.f;
            if (mrow < NC_) {
                w0 = nw[mrow * 64 + k];
                w1 = nw[mrow * 64 + 32 + k];
            }
            a0.h[e] = (_Float16)w0;
            a1.h[e] = (_Float16)w1;
        }
        // Per-D-register bias (D layout: VGPR r holds M = base + r + 8*half)
        float bias[8];
        const int mb = mt * 16 + 8 * half;
#pragma unroll
        for (int r = 0; r < 8; ++r) {
            const int m = mb + r;
            bias[r] = (m < NC_) ? nb[m] : 0.f;
        }

        float rowsum[8];
#pragma unroll
        for (int r = 0; r < 8; ++r) rowsum[r] = 0.f;

        for (int nt = 0; nt < 8; ++nt) {
            // B fragments (32x16 f16): lane l -> N = nt*16 + (l&15);
            // K = 16*(l>>4) + e. K-contiguous in xh -> one 32B vector load each.
            const int tcol = nt * 16 + mloc;
            const int kb   = 16 * half;
            const _Float16* xrow = &xh[tcol * HSTRIDE];
            const v16h bv0 = *(const v16h*)(xrow + kb);        // K block [0,32)
            const v16h bv1 = *(const v16h*)(xrow + 32 + kb);   // K block [32,64)

            FragD acc; acc.v = (v8f){};
            acc.v = __builtin_amdgcn_wmma_f32_16x16x32_f16(
                false, a0.v, false, bv0, (short)0, acc.v, false, false);
            acc.v = __builtin_amdgcn_wmma_f32_16x16x32_f16(
                false, a1.v, false, bv1, (short)0, acc.v, false, false);
            // bias + clip (per element, pre-pooling), accumulate over N
#pragma unroll
            for (int r = 0; r < 8; ++r) {
                float v = acc.f[r] + bias[r];
                v = fminf(fmaxf(v, 0.f), 1.f);
                rowsum[r] += v;
            }
        }

        // Reduce over the 16 N-lanes of each half; lane0 -> M=base+r, lane16 -> M=base+r+8
#pragma unroll
        for (int r = 0; r < 8; ++r) {
            float v = rowsum[r];
#pragma unroll
            for (int off = 1; off < 16; off <<= 1)
                v += __shfl_xor(v, off, 32);
            if ((lane & 15) == 0) {
                const int m = mt * 16 + r + 8 * half;
                if (m < NC_)
                    spec[((size_t)b * NC_ + m) * CHUNKS_ + j] = v * (1.f / 128.f);
            }
        }
    }
}

// =====================================================================
// Kernel 2: per (batch, frame s): rfft(256,'ortho') of the white-noise frame,
// multiply by spec[b,:,s], irfft(256,'ortho'), overlap-add into out with
// noise_factor. Twiddles via a 256-entry LDS table (phase = k*n mod 256).
// grid = 32*128 blocks, 256 threads.
// =====================================================================
__global__ __launch_bounds__(256) void ddsp_noise_bank(
    const float* __restrict__ white,  // (32, 16384)
    const float* __restrict__ spec,   // (32, 129, 128)
    const float* __restrict__ nfac,   // scalar
    float* __restrict__ out)          // (32, 16384), pre-filled with channel mean
{
    __shared__ float frame[256];
    __shared__ float ctab[256], stab[256];
    __shared__ float fre[NC_], fim[NC_];

    const int blk = blockIdx.x;
    const int b   = blk >> 7;
    const int s   = blk & 127;
    const int tid = threadIdx.x;

    const float ang = 6.28318530718f * (float)tid * (1.f / 256.f);
    ctab[tid] = __cosf(ang);
    stab[tid] = __sinf(ang);
    const int idx = s * 128 + tid;                      // frame window (zero-padded tail)
    frame[tid] = (idx < BAND_) ? white[(size_t)b * BAND_ + idx] : 0.f;
    __syncthreads();

    // forward DFT bins 0..128, scaled by spec (ortho norms folded into final 1/256)
    if (tid <= 128) {
        float re = 0.f, im = 0.f;
        for (int n = 0; n < 256; ++n) {
            const int ph = (tid * n) & 255;
            const float f = frame[n];
            re += f * ctab[ph];
            im -= f * stab[ph];
        }
        const float sp = spec[((size_t)b * NC_ + tid) * CHUNKS_ + s];
        fre[tid] = sp * re;
        fim[tid] = sp * im;
    }
    __syncthreads();

    // inverse DFT sample tid in [0,256)
    const float scale = nfac[0] * (1.f / 256.f);
    float acc = fre[0] + fre[128] * ((tid & 1) ? -1.f : 1.f);
    for (int k = 1; k < 128; ++k) {
        const int ph = (k * tid) & 255;
        acc += 2.f * (fre[k] * ctab[ph] - fim[k] * stab[ph]);
    }
    acc *= scale;

    // overlap-add: head -> this chunk; tail -> next chunk (tail of last frame dropped)
    if (tid < 128) {
        atomicAdd(out + (size_t)b * BAND_ + s * 128 + tid, acc);
    } else if (s < 127) {
        atomicAdd(out + (size_t)b * BAND_ + (s + 1) * 128 + (tid - 128), acc);
    }
}

// =====================================================================
extern "C" void kernel_launch(void* const* d_in, const int* in_sizes, int n_in,
                              void* d_out, int out_size, void* d_ws, size_t ws_size,
                              hipStream_t stream) {
    const float* x       = (const float*)d_in[0];   // (32,64,16384)
    // d_in[1..4]: amp_w/amp_b/freq_w/freq_b -> dead code in reference output
    const float* noise_w = (const float*)d_in[5];   // (129,64)
    const float* noise_b = (const float*)d_in[6];   // (129,)
    const float* nfac    = (const float*)d_in[7];   // scalar
    const float* white   = (const float*)d_in[8];   // (32,16384)
    float* out  = (float*)d_out;                    // (32,1,16384)
    float* spec = (float*)d_ws;                     // (32,129,128) fp32 = ~2.1 MB

    dim3 grid(B_ * CHUNKS_), block(256);
    ddsp_gemm_mean<<<grid, block, 0, stream>>>(x, noise_w, noise_b, out, spec);
    ddsp_noise_bank<<<grid, block, 0, stream>>>(white, spec, nfac, out);
}
